// Attention_52115133170131
// MI455X (gfx1250) — compile-verified, matrix operand order checked by reading, not run on hardware
//
#include <hip/hip_runtime.h>
#include <hip/hip_bf16.h>

// ---------------------------------------------------------------------------
// Types for WMMA (trivially copyable so they can live in unions)
// ---------------------------------------------------------------------------
typedef __bf16        bf16x16 __attribute__((ext_vector_type(16)));
typedef float         f32x8   __attribute__((ext_vector_type(8)));
typedef unsigned int  u32x4   __attribute__((ext_vector_type(4)));
typedef int           i32x4   __attribute__((ext_vector_type(4)));
typedef int           i32x8   __attribute__((ext_vector_type(8)));

union Frag {
    u32x4   q[2];   // 8 dwords = 16 bf16
    bf16x16 v;
};

__device__ __forceinline__ unsigned short f2bf(float f) {
    unsigned int u = __float_as_uint(f);
    unsigned int r = u + 0x7FFFu + ((u >> 16) & 1u);   // round-to-nearest-even
    return (unsigned short)(r >> 16);
}

#define HD 128          // head dim

// ---------------------------------------------------------------------------
// CDNA5 async global->LDS copy (ASYNCcnt-tracked). IOFFSET is added to BOTH
// the LDS and global addresses (CDNA5 ISA), so one LDS-base
// VGPR + one global-base VGPR serves consecutive 16B chunks.
// ---------------------------------------------------------------------------
#define ASYNC_LDS_B128(ldsaddr, gaddr, off)                                   \
    asm volatile("global_load_async_to_lds_b128 %0, %1, off offset:" #off     \
                 :: "v"(ldsaddr), "v"(gaddr) : "memory")

#define WAIT_ASYNCCNT(n)                                                      \
    asm volatile("s_wait_asynccnt " #n ::: "memory")

// Tensor Data Mover availability (this toolchain: 6-arg builtin,
// amdgpu-toolchain clang-23 form per bridge doc)
#if defined(__gfx1250__) && __has_builtin(__builtin_amdgcn_tensor_load_to_lds) \
                         && __has_builtin(__builtin_amdgcn_s_wait_tensorcnt)
#define HAVE_TDM 1
#else
#define HAVE_TDM 0
#endif

// ---------------------------------------------------------------------------
// fp32 -> bf16 elementwise conversion (grid-stride)
// ---------------------------------------------------------------------------
__global__ void cvt_bf16_kernel(const float* __restrict__ in,
                                unsigned short* __restrict__ out, size_t n) {
    size_t i = (size_t)blockIdx.x * blockDim.x + threadIdx.x;
    size_t stride = (size_t)gridDim.x * blockDim.x;
    for (; i < n; i += stride) out[i] = f2bf(in[i]);
}

// ---------------------------------------------------------------------------
// WMMA GEMM:  C[M,N] (f32) = A[M,K] (bf16, row-major) * B[N,K]^T (bf16)
// 128x128 tile / workgroup, 8 waves; each wave 64x32 = 4x2 WMMA tiles (8 WMMAs
// per K-step of 32). Double-buffered LDS fed by async global->LDS copies: the
// next tile streams in (ASYNCcnt) while WMMAs run on the current tile.
// ---------------------------------------------------------------------------
#define GT_M 128
#define GT_N 128
#define GT_K 32
#define LDT  40   // padded LDS row stride in elements (80B, multiple of 16B)

__global__ __launch_bounds__(256)
void gemm_bf16_nt_kernel(const unsigned short* __restrict__ A,
                         const unsigned short* __restrict__ Bm,
                         float* __restrict__ C,
                         int M, int N, int K) {
    __shared__ __attribute__((aligned(16))) unsigned short As[2][GT_M * LDT];
    __shared__ __attribute__((aligned(16))) unsigned short Bs[2][GT_N * LDT];

    const int tid  = threadIdx.x;
    const int lane = tid & 31;
    const int wave = tid >> 5;            // 0..7
    const int m0 = blockIdx.y * GT_M;
    const int n0 = blockIdx.x * GT_N;
    const int wr = (wave >> 2) * 64;      // wave row offset inside tile (0,64)
    const int wc = (wave & 3) * 32;       // wave col offset inside tile
    const int half = lane >> 4;
    const int l15  = lane & 15;

    const int lrow = tid >> 1;            // 0..127 (loader row)
    const int lcol = (tid & 1) * 16;      // 0 or 16 (loader col, elements)

    const size_t arow = (size_t)(m0 + lrow) * K + lcol;
    const size_t brow = (size_t)(n0 + lrow) * K + lcol;
    const unsigned ldsA0 = (unsigned)(size_t)(&As[0][lrow * LDT + lcol]);
    const unsigned ldsA1 = (unsigned)(size_t)(&As[1][lrow * LDT + lcol]);
    const unsigned ldsB0 = (unsigned)(size_t)(&Bs[0][lrow * LDT + lcol]);
    const unsigned ldsB1 = (unsigned)(size_t)(&Bs[1][lrow * LDT + lcol]);

    auto issue_tile = [&](int buf, int k0) {
        unsigned la = buf ? ldsA1 : ldsA0;
        unsigned lb = buf ? ldsB1 : ldsB0;
        unsigned long long ga = (unsigned long long)(size_t)(A  + arow + k0);
        unsigned long long gb = (unsigned long long)(size_t)(Bm + brow + k0);
        ASYNC_LDS_B128(la, ga, 0);
        ASYNC_LDS_B128(la, ga, 16);
        ASYNC_LDS_B128(lb, gb, 0);
        ASYNC_LDS_B128(lb, gb, 16);
    };

    f32x8 acc[4][2] = {};

    const int nkt = K / GT_K;
    issue_tile(0, 0);                     // prologue: tile 0 in flight

    for (int kt = 0; kt < nkt; ++kt) {
        const int buf = kt & 1;
        if (kt + 1 < nkt) {
            issue_tile(buf ^ 1, (kt + 1) * GT_K);   // stream next tile
            WAIT_ASYNCCNT(0x4);                     // current tile's 4 ops done
        } else {
            WAIT_ASYNCCNT(0x0);
        }
        if (kt + 2 < nkt) {                         // L2 prefetch 2 tiles ahead
            __builtin_prefetch(A  + arow + (size_t)(kt + 2) * GT_K, 0, 0);
            __builtin_prefetch(Bm + brow + (size_t)(kt + 2) * GT_K, 0, 0);
        }
        __syncthreads();                            // other waves' copies done

        const unsigned short* Ab = As[buf];
        const unsigned short* Bb = Bs[buf];
        Frag af[4], bfr[2];
        #pragma unroll
        for (int t = 0; t < 4; ++t) {
            const unsigned short* ar = Ab + (wr + t * 16 + l15) * LDT;
            af[t].q[0]  = *(const u32x4*)(ar + half * 8);
            af[t].q[1]  = *(const u32x4*)(ar + 16 + half * 8);
        }
        #pragma unroll
        for (int t = 0; t < 2; ++t) {
            const unsigned short* br = Bb + (wc + t * 16 + l15) * LDT;
            bfr[t].q[0] = *(const u32x4*)(br + half * 8);
            bfr[t].q[1] = *(const u32x4*)(br + 16 + half * 8);
        }
        #pragma unroll
        for (int i = 0; i < 4; ++i)
            #pragma unroll
            for (int j = 0; j < 2; ++j)
                acc[i][j] = __builtin_amdgcn_wmma_f32_16x16x32_bf16(
                    false, af[i].v, false, bfr[j].v, (short)0, acc[i][j],
                    false, false);
        __syncthreads();   // everyone done reading buf before it is refilled
    }

    // Epilogue: C layout -> lane element (r): row = r + 8*half, col = l15
    #pragma unroll
    for (int i = 0; i < 4; ++i)
        #pragma unroll
        for (int j = 0; j < 2; ++j)
            #pragma unroll
            for (int r = 0; r < 8; ++r) {
                int row = m0 + wr + i * 16 + r + 8 * half;
                int col = n0 + wc + j * 16 + l15;
                C[(size_t)row * N + col] = acc[i][j][r];
            }
}

// ---------------------------------------------------------------------------
// RoPE: X (B,S,H,D) f32 -> Y (B,H,S,D) bf16 (rotated); optional pre-rope f32
// copy P (B,H,S,D) for new_k.  One thread per pair.
// ---------------------------------------------------------------------------
__global__ void rope_kernel(const float* __restrict__ X,
                            unsigned short* __restrict__ Y,
                            float* __restrict__ P,
                            int B, int S, int H) {
    size_t i = (size_t)blockIdx.x * blockDim.x + threadIdx.x;
    size_t total = (size_t)B * S * H * (HD / 2);
    if (i >= total) return;
    int p = (int)(i % (HD / 2)); size_t t = i / (HD / 2);
    int h = (int)(t % H); t /= H;
    int s = (int)(t % S); int b = (int)(t / S);

    float inv = __expf(-((float)(2 * p) / (float)HD) * 9.2103403720f); // ln(1e4)
    float ang = (float)s * inv;
    float sn, cs;
    __sincosf(ang, &sn, &cs);

    size_t src = ((size_t)(b * S + s) * H + h) * HD;
    float x1 = X[src + 2 * p];
    float x2 = X[src + 2 * p + 1];
    size_t dst = ((size_t)(b * H + h) * S + s) * HD;
    Y[dst + 2 * p]     = f2bf(x1 * cs - x2 * sn);
    Y[dst + 2 * p + 1] = f2bf(x1 * sn + x2 * cs);
    if (P) { P[dst + 2 * p] = x1; P[dst + 2 * p + 1] = x2; }
}

// ---------------------------------------------------------------------------
// V pass: X (B,S,H,D) f32 -> Y (B,H,S,D) bf16 + P (B,H,S,D) f32 (new_v)
// ---------------------------------------------------------------------------
__global__ void v_pack_kernel(const float* __restrict__ X,
                              unsigned short* __restrict__ Y,
                              float* __restrict__ P,
                              int B, int S, int H) {
    size_t i = (size_t)blockIdx.x * blockDim.x + threadIdx.x;
    size_t total = (size_t)B * S * H * HD;
    if (i >= total) return;
    int d = (int)(i % HD); size_t t = i / HD;
    int h = (int)(t % H); t /= H;
    int s = (int)(t % S); int b = (int)(t / S);
    float v = X[((size_t)(b * S + s) * H + h) * HD + d];
    size_t dst = ((size_t)(b * H + h) * S + s) * HD + d;
    Y[dst] = f2bf(v);
    P[dst] = v;
}

// ---------------------------------------------------------------------------
// Flash attention, causal. 4 waves / workgroup; each wave owns 16 query rows
// (64 rows per block). The K tile (32 keys x 128) is double-buffered in LDS
// and fetched by the Tensor Data Mover (TENSORcnt): wave 0 issues the TDM for
// block kb+1, then s_wait_tensorcnt 1 guarantees block kb is resident (TDM is
// in-order per wave) before the workgroup barrier.
// V is staged transposed; P is re-laid-out through per-wave LDS.
// ---------------------------------------------------------------------------
__global__ __launch_bounds__(128)
void flash_attn_kernel(const unsigned short* __restrict__ Q,   // (B,H,S,D) bf16
                       const unsigned short* __restrict__ Kk,  // (B,KVH,S,D)
                       const unsigned short* __restrict__ V,   // (B,KVH,S,D)
                       unsigned short* __restrict__ O,         // (B,S,H*D) bf16
                       int S, int H, int KVH) {
    __shared__ __attribute__((aligned(16))) unsigned short Ks[2][32 * HD];   // [key][d]
    __shared__ __attribute__((aligned(16))) unsigned short Vt[HD * 32];      // [d][key]
    __shared__ __attribute__((aligned(16))) unsigned short Pl[4][16 * 32];   // per-wave P

    const int tid  = threadIdx.x;
    const int lane = tid & 31;
    const int w    = tid >> 5;
    const int half = lane >> 4;
    const int l15  = lane & 15;
    const int h = blockIdx.y, b = blockIdx.z;
    const int q0 = blockIdx.x * 64;          // block's first query row
    const int qw = q0 + w * 16;              // this wave's first query row
    const int kvh = h / (H / KVH);
    const size_t kvbase = (size_t)(b * KVH + kvh) * S * HD;
    const int nkb = (q0 + 64) >> 5;          // key blocks for the whole block

#if HAVE_TDM
    // Tensor DMA descriptor (D#, CDNA5 ISA):
    // 2-D tensor: (S rows x 128 cols, 2B elems); tile = 32 rows x 128 cols.
    auto tdm_issue_k = [&](const unsigned short* dstLds, int krow) {
        unsigned long long ga =
            (unsigned long long)(size_t)(Kk + kvbase + (size_t)krow * HD);
        unsigned lds = (unsigned)(size_t)dstLds;
        u32x4 g0;
        g0[0] = 1u;                                  // count=1 (valid user D#)
        g0[1] = lds;                                 // lds_addr (bytes)
        g0[2] = (unsigned)ga;                        // global_addr[31:0]
        g0[3] = (unsigned)(ga >> 32) | (2u << 30);   // global_addr[56:32], type=2
        i32x8 g1;
        g1[0] = (int)(1u << 16);                     // data_size=1 (2 bytes)
        g1[1] = (int)(HD << 16);                     // tensor_dim0 = 128
        g1[2] = (int)((S & 0xffff) << 16);           // tensor_dim1[15:0]
        g1[3] = (int)(HD << 16);                     // tile_dim0 = 128 | dim1 hi=0
        g1[4] = 32;                                  // tile_dim1 = 32, tile_dim2=0
        g1[5] = HD;                                  // tensor_dim0_stride = 128
        int sd1 = S * HD;                            // tensor_dim1_stride
        g1[6] = (int)((sd1 & 0xffff) << 16);
        g1[7] = (int)(((unsigned)sd1) >> 16);
        i32x4 z4 = {0, 0, 0, 0};
        i32x8 z8 = {0, 0, 0, 0, 0, 0, 0, 0};
        __builtin_amdgcn_tensor_load_to_lds(g0, g1, z4, z4, z8, 0);
    };
#endif

    // Q fragments: 4 x (16x32) along head dim
    const size_t qrow = ((size_t)(b * H + h) * S + qw + l15) * HD;
    Frag qf[4];
    #pragma unroll
    for (int kd = 0; kd < 4; ++kd) {
        qf[kd].q[0] = *(const u32x4*)(Q + qrow + kd * 32 + half * 8);
        qf[kd].q[1] = *(const u32x4*)(Q + qrow + kd * 32 + 16 + half * 8);
    }

    f32x8 acc[8] = {};
    float m_i[8], l_i[8];
    #pragma unroll
    for (int r = 0; r < 8; ++r) { m_i[r] = -3.0e38f; l_i[r] = 0.f; }

    const float scale = 0.08838834764831845f;  // 1/sqrt(128)

#if HAVE_TDM
    if (w == 0) tdm_issue_k(&Ks[0][0], 0);     // prologue: K tile 0 in flight
#endif

    for (int kb = 0; kb < nkb; ++kb) {
        const int k0 = kb * 32;
        const int kbuf = kb & 1;
#if HAVE_TDM
        if (w == 0) {
            if (kb + 1 < nkb) {
                tdm_issue_k(&Ks[kbuf ^ 1][0], k0 + 32);   // stream next K tile
                __builtin_amdgcn_s_wait_tensorcnt(1);     // current tile done
            } else {
                __builtin_amdgcn_s_wait_tensorcnt(0);
            }
        }
#else
        {   // cooperative K tile load: 32 rows x 128 d (8KB)
            const int row = tid >> 2, seg = (tid & 3) * 32;
            const u32x4* g = (const u32x4*)(Kk + kvbase + (size_t)(k0 + row) * HD + seg);
            u32x4* d = (u32x4*)(&Ks[kbuf][row * HD + seg]);
            d[0] = g[0]; d[1] = g[1]; d[2] = g[2]; d[3] = g[3];
        }
#endif
        // ---- cooperative V tile load, transposed into LDS: Vt[d][kk]
        {
            const int kk = tid >> 2, dseg = (tid & 3) * 32;
            const unsigned short* g = V + kvbase + (size_t)(k0 + kk) * HD + dseg;
            #pragma unroll 8
            for (int d = 0; d < 32; ++d)
                Vt[(dseg + d) * 32 + kk] = g[d];
        }
        __syncthreads();

        if (k0 < qw + 16) {        // wave-uniform causal early-out
            const unsigned short* Kb = &Ks[kbuf][0];
            // ---- scores: two 16x16 tiles over 32 keys (K frags from LDS)
            f32x8 s[2] = {};
            #pragma unroll
            for (int t = 0; t < 2; ++t) {
                const unsigned short* kr = Kb + (t * 16 + l15) * HD;
                #pragma unroll
                for (int kd = 0; kd < 4; ++kd) {
                    Frag kf;
                    kf.q[0] = *(const u32x4*)(kr + kd * 32 + half * 8);
                    kf.q[1] = *(const u32x4*)(kr + kd * 32 + 16 + half * 8);
                    s[t] = __builtin_amdgcn_wmma_f32_16x16x32_bf16(
                        false, qf[kd].v, false, kf.v, (short)0, s[t], false, false);
                }
            }
            // ---- scale + causal mask
            #pragma unroll
            for (int t = 0; t < 2; ++t)
                #pragma unroll
                for (int r = 0; r < 8; ++r) {
                    int row = qw + r + 8 * half;
                    int col = k0 + t * 16 + l15;
                    float sv = s[t][r] * scale;
                    s[t][r] = (col <= row) ? sv : -3.0e38f;
                }
            // ---- online softmax (row stats replicated across 16-lane halves)
            float alpha[8];
            #pragma unroll
            for (int r = 0; r < 8; ++r) {
                float mx = fmaxf(s[0][r], s[1][r]);
                #pragma unroll
                for (int m = 1; m < 16; m <<= 1) mx = fmaxf(mx, __shfl_xor(mx, m, 32));
                float mn = fmaxf(m_i[r], mx);
                alpha[r] = __expf(m_i[r] - mn);
                m_i[r] = mn;
                float p0 = __expf(s[0][r] - mn);
                float p1 = __expf(s[1][r] - mn);
                s[0][r] = p0; s[1][r] = p1;
                float sum = p0 + p1;
                #pragma unroll
                for (int m = 1; m < 16; m <<= 1) sum += __shfl_xor(sum, m, 32);
                l_i[r] = l_i[r] * alpha[r] + sum;
            }
            #pragma unroll
            for (int j = 0; j < 8; ++j)
                #pragma unroll
                for (int r = 0; r < 8; ++r)
                    acc[j][r] *= alpha[r];

            // ---- P (C-layout) -> per-wave LDS bf16 (same-wave DS is in-order)
            unsigned short* pw = Pl[w];
            #pragma unroll
            for (int t = 0; t < 2; ++t)
                #pragma unroll
                for (int r = 0; r < 8; ++r)
                    pw[(r + 8 * half) * 32 + t * 16 + l15] = f2bf(s[t][r]);

            // ---- O += P * V : P as A-frag (16x32), V as 8 B-frags (32x16)
            Frag pf;
            pf.q[0] = *(const u32x4*)(pw + l15 * 32 + half * 8);
            pf.q[1] = *(const u32x4*)(pw + l15 * 32 + 16 + half * 8);
            #pragma unroll
            for (int j = 0; j < 8; ++j) {
                Frag vf;
                const unsigned short* vtr = Vt + (j * 16 + l15) * 32;
                vf.q[0] = *(const u32x4*)(vtr + half * 8);
                vf.q[1] = *(const u32x4*)(vtr + 16 + half * 8);
                acc[j] = __builtin_amdgcn_wmma_f32_16x16x32_bf16(
                    false, pf.v, false, vf.v, (short)0, acc[j], false, false);
            }
        }
        __syncthreads();   // all waves done with Ks/Vt before refill
    }

    // ---- normalize + store bf16 into (B,S,H*D) for the out-projection GEMM
    #pragma unroll
    for (int j = 0; j < 8; ++j)
        #pragma unroll
        for (int r = 0; r < 8; ++r) {
            int row = qw + r + 8 * half;
            int d = j * 16 + l15;
            float val = acc[j][r] / l_i[r];
            O[((size_t)(b * S + row) * H + h) * HD + d] = f2bf(val);
        }
}

// ---------------------------------------------------------------------------
// Host-side orchestration
// ---------------------------------------------------------------------------
extern "C" void kernel_launch(void* const* d_in, const int* in_sizes, int n_in,
                              void* d_out, int out_size, void* d_ws, size_t ws_size,
                              hipStream_t stream) {
    (void)in_sizes; (void)n_in; (void)out_size; (void)ws_size;

    constexpr int B = 2, S = 2048, DIM = 4096, H = 32, KVH = 8, D = HD;
    constexpr int M = B * S;                  // 4096 token rows
    constexpr size_t NQ  = (size_t)H   * D;   // 4096
    constexpr size_t NKV = (size_t)KVH * D;   // 1024

    const float* x  = (const float*)d_in[0];
    const float* wq = (const float*)d_in[1];
    const float* wk = (const float*)d_in[2];
    const float* wv = (const float*)d_in[3];
    const float* wo = (const float*)d_in[4];

    float* out_f  = (float*)d_out;                          // (B,S,DIM)
    float* new_k  = out_f + (size_t)M * DIM;                // (B,KVH,S,D)
    float* new_v  = new_k + (size_t)B * KVH * S * D;        // (B,KVH,S,D)

    // Workspace carve-up
    char* p = (char*)d_ws;
    auto takeU16 = [&](size_t n) { unsigned short* r = (unsigned short*)p; p += n * 2; return r; };
    auto takeF32 = [&](size_t n) { float* r = (float*)p; p += n * 4; return r; };

    unsigned short* x_bf    = takeU16((size_t)M * DIM);
    unsigned short* wq_bf   = takeU16(NQ  * DIM);
    unsigned short* wk_bf   = takeU16(NKV * DIM);
    unsigned short* wv_bf   = takeU16(NKV * DIM);
    unsigned short* wo_bf   = takeU16((size_t)DIM * NQ);
    float*          xq_f    = takeF32((size_t)M * NQ);
    float*          xk_f    = takeF32((size_t)M * NKV);
    float*          xv_f    = takeF32((size_t)M * NKV);
    unsigned short* q_bf    = takeU16((size_t)M * NQ);    // (B,H,S,D)
    unsigned short* k_bf    = takeU16((size_t)M * NKV);   // (B,KVH,S,D)
    unsigned short* v_bf    = takeU16((size_t)M * NKV);   // (B,KVH,S,D)
    unsigned short* attn_bf = takeU16((size_t)M * NQ);    // (B,S,H*D)

    // 1) fp32 -> bf16 conversions
    const int CT = 256, CG = 4096;
    cvt_bf16_kernel<<<CG, CT, 0, stream>>>(x,  x_bf,  (size_t)M * DIM);
    cvt_bf16_kernel<<<CG, CT, 0, stream>>>(wq, wq_bf, NQ  * DIM);
    cvt_bf16_kernel<<<CG, CT, 0, stream>>>(wk, wk_bf, NKV * DIM);
    cvt_bf16_kernel<<<CG, CT, 0, stream>>>(wv, wv_bf, NKV * DIM);
    cvt_bf16_kernel<<<CG, CT, 0, stream>>>(wo, wo_bf, (size_t)DIM * NQ);

    // 2) QKV projections (C = A * B^T), async-pipelined WMMA
    gemm_bf16_nt_kernel<<<dim3(NQ  / GT_N, M / GT_M), 256, 0, stream>>>(
        x_bf, wq_bf, xq_f, M, (int)NQ, DIM);
    gemm_bf16_nt_kernel<<<dim3(NKV / GT_N, M / GT_M), 256, 0, stream>>>(
        x_bf, wk_bf, xk_f, M, (int)NKV, DIM);
    gemm_bf16_nt_kernel<<<dim3(NKV / GT_N, M / GT_M), 256, 0, stream>>>(
        x_bf, wv_bf, xv_f, M, (int)NKV, DIM);

    // 3) RoPE on q and k (k also emits pre-RoPE new_k); v pack (emits new_v)
    {
        size_t tq = (size_t)B * S * H * (D / 2);
        rope_kernel<<<(unsigned)((tq + 255) / 256), 256, 0, stream>>>(
            xq_f, q_bf, nullptr, B, S, H);
        size_t tk = (size_t)B * S * KVH * (D / 2);
        rope_kernel<<<(unsigned)((tk + 255) / 256), 256, 0, stream>>>(
            xk_f, k_bf, new_k, B, S, KVH);
        size_t tv = (size_t)B * S * KVH * D;
        v_pack_kernel<<<(unsigned)((tv + 255) / 256), 256, 0, stream>>>(
            xv_f, v_bf, new_v, B, S, KVH);
    }

    // 4) causal flash attention -> attn_bf (B,S,H*D)
    flash_attn_kernel<<<dim3(S / 64, H, B), 128, 0, stream>>>(
        q_bf, k_bf, v_bf, attn_bf, S, H, KVH);

    // 5) output projection: out = attn @ wo^T  (fp32 result straight to d_out)
    gemm_bf16_nt_kernel<<<dim3(DIM / GT_N, M / GT_M), 256, 0, stream>>>(
        attn_bf, wo_bf, out_f, M, DIM, DIM);
}
